// Net_25847113187879
// MI455X (gfx1250) — matrix operand order repaired and checked by reading, not compiled, not run on hardware
//
#include <hip/hip_runtime.h>
#include <hip/hip_bf16.h>

typedef __attribute__((ext_vector_type(16))) _Float16 v16h;
typedef __attribute__((ext_vector_type(8)))  _Float16 v8h;
typedef __attribute__((ext_vector_type(8)))  float    v8f;

#define BTOT   131072
#define DAYMAX 256
#define NGRP   (BTOT / DAYMAX)   // 512
#define HOLDN  60

__device__ __forceinline__ float leaky_(float v) { return v >= 0.f ? v : 0.1f * v; }

// ---------------------------------------------------------------------------
// Kernel A: build f16 feature rows yA (from x) and yB (from xp), padded to 96.
// Layout: [0..7]=c1, [8..39]=cr, [40..49]=base, [50..56]=onehot,
//         [57..67]=mirr, [68..78]=nomirr, [79..95]=0 (sh_price filled later).
// ---------------------------------------------------------------------------
__global__ void features_kernel(const float* __restrict__ x, const float* __restrict__ xp,
                                const float* __restrict__ price, const float* __restrict__ nom,
                                const int* __restrict__ cat,
                                const float* __restrict__ c00w, const float* __restrict__ c00b,
                                const float* __restrict__ c10w, const float* __restrict__ c10b,
                                const float* __restrict__ oh00, const float* __restrict__ oh01,
                                const float* __restrict__ oh02, const float* __restrict__ oh03,
                                const float* __restrict__ oh04,
                                _Float16* __restrict__ yA, _Float16* __restrict__ yB)
{
    long r = (long)blockIdx.x * blockDim.x + threadIdx.x;
    if (r >= BTOT) return;

    float fa[96], fb[96];
    const float* vx = x  + r * 60;
    const float* vp = xp + r * 60;

    // conv00 on slice 0 -> c1 (8)
    #pragma unroll
    for (int o = 0; o < 8; ++o) {
        float sa = c00b[o], sb = c00b[o];
        #pragma unroll
        for (int ck = 0; ck < 10; ++ck) {   // c*5+k
            float w = c00w[o * 10 + ck];
            sa += vx[ck] * w;
            sb += vp[ck] * w;
        }
        fa[o] = leaky_(sa); fb[o] = leaky_(sb);
    }
    // conv10 on slices 1..4 -> cr (4x8)
    #pragma unroll
    for (int i = 0; i < 4; ++i) {
        const float* sx = vx + (1 + i) * 10;
        const float* sp = vp + (1 + i) * 10;
        #pragma unroll
        for (int o = 0; o < 8; ++o) {
            float sa = c10b[o], sb = c10b[o];
            #pragma unroll
            for (int ck = 0; ck < 10; ++ck) {
                float w = c10w[o * 10 + ck];
                sa += sx[ck] * w;
                sb += sp[ck] * w;
            }
            fa[8 + i * 8 + o] = leaky_(sa); fb[8 + i * 8 + o] = leaky_(sb);
        }
    }
    // base = slice 5 flattened (10)
    #pragma unroll
    for (int j = 0; j < 10; ++j) { fa[40 + j] = vx[50 + j]; fb[40 + j] = vp[50 + j]; }

    // onehot (7):  m4 only affects col 1 of oh04
    int c0 = cat[r * 5 + 0], c1 = cat[r * 5 + 1], c2 = cat[r * 5 + 2],
        c3 = cat[r * 5 + 3], c4 = cat[r * 5 + 4];
    float m41 = 0.f;
    #pragma unroll
    for (int i = 0; i < 10; ++i) m41 += oh04[i * 2 + 1];
    m41 *= 0.1f;
    float oh[7];
    oh[0] = oh00[c0]; oh[1] = oh01[c1]; oh[2] = oh02[c2];
    oh[3] = oh03[c3 * 2]; oh[4] = oh03[c3 * 2 + 1];
    oh[5] = oh04[c4 * 2]; oh[6] = oh04[c4 * 2 + 1] - m41;
    #pragma unroll
    for (int i = 0; i < 7; ++i) { fa[50 + i] = oh[i]; fb[50 + i] = oh[i]; }
    fb[56] = -oh[6];                      // onehot_neg: last col negated

    // mirr (11) and nomirr (11)
    #pragma unroll
    for (int i = 0; i < 11; ++i) {
        float pv = price[r * 12 + 1 + i];
        fa[57 + i] = pv; fb[57 + i] = -pv;
    }
    #pragma unroll
    for (int i = 0; i < 11; ++i) {
        float nv = nom[r * 13 + 2 + i];
        fa[68 + i] = nv; fb[68 + i] = nv;
    }
    #pragma unroll
    for (int i = 79; i < 96; ++i) { fa[i] = 0.f; fb[i] = 0.f; }

    _Float16* oa = yA + r * 96;
    _Float16* ob = yB + r * 96;
    #pragma unroll
    for (int i = 0; i < 96; ++i) { oa[i] = (_Float16)fa[i]; ob[i] = (_Float16)fb[i]; }
}

// ---------------------------------------------------------------------------
// WMMA GEMM: out[B x N] = relu(A[B x KP](f16) * W^T + bias)
// W (row-major N x Kraw, f32) is staged in LDS as f16 *pre-swizzled into
// B-fragment layout*: for tile (kt,nt), lane L's 16 contiguous halves live at
// Wlds[((kt*NT+nt)*32 + L)*16 + i]  ==  W[kt*32 + (L>>4)*16 + i][nt*16 + (L&15)]
// so each B fragment is two ds_load_b128 ops (no scalar u16 gathers).
// One wave handles 16 rows; 8 waves / 256-thread block -> 128 rows per block.
// ---------------------------------------------------------------------------
template <int KP, int N, bool OUTH>
__global__ void gemm_relu(const _Float16* __restrict__ A,
                          const float* __restrict__ Wg, int Kraw,
                          const float* __restrict__ bias,
                          _Float16* __restrict__ outH, float* __restrict__ outF)
{
    constexpr int NT = N / 16;
    constexpr int KT = KP / 32;
    __shared__ _Float16 Wlds[KP * N];   // fragment-layout, fully packed

    for (int i = threadIdx.x; i < KP * N; i += blockDim.x) {
        int k = i / N, n = i % N;
        float w = (k < Kraw) ? Wg[n * Kraw + k] : 0.f;
        int kt = k >> 5, kr = k & 31;
        int fhi = kr >> 4, ii = kr & 15;
        int nt = n >> 4, col = n & 15;
        int lane = fhi * 16 + col;
        Wlds[(((kt * NT + nt) * 32 + lane) << 4) + ii] = (_Float16)w;
    }
    __syncthreads();

    int wave = threadIdx.x >> 5;
    int lane = threadIdx.x & 31;
    int mrow = lane & 15;
    int hi   = lane >> 4;
    long m0  = ((long)blockIdx.x * (blockDim.x >> 5) + wave) * 16;
    const _Float16* Arow = A + (m0 + mrow) * KP;
    const _Float16* Wlane = Wlds + (lane << 4);

    v8f acc[NT];
    #pragma unroll
    for (int nt = 0; nt < NT; ++nt)
        #pragma unroll
        for (int i = 0; i < 8; ++i) acc[nt][i] = 0.f;

    #pragma unroll
    for (int kt = 0; kt < KT; ++kt) {
        // A fragment: lane holds K = {kb..kb+15}, kb = 32kt + 16*hi
        int kb = kt * 32 + hi * 16;
        v8h alo = *(const v8h*)(Arow + kb);
        v8h ahi = *(const v8h*)(Arow + kb + 8);
        v16h a = __builtin_shufflevector(alo, ahi,
                                         0, 1, 2, 3, 4, 5, 6, 7,
                                         8, 9, 10, 11, 12, 13, 14, 15);
        #pragma unroll
        for (int nt = 0; nt < NT; ++nt) {
            const _Float16* bp = Wlane + ((kt * NT + nt) << 9);  // *32*16
            v8h blo = *(const v8h*)(bp);
            v8h bhi = *(const v8h*)(bp + 8);
            v16h b = __builtin_shufflevector(blo, bhi,
                                             0, 1, 2, 3, 4, 5, 6, 7,
                                             8, 9, 10, 11, 12, 13, 14, 15);
            acc[nt] = __builtin_amdgcn_wmma_f32_16x16x32_f16(
                false, a, false, b, (short)0, acc[nt], false, false);
        }
    }

    // bias + relu + store  (D: VGPR r -> M = r + 8*hi, N = 16*nt + mrow)
    #pragma unroll
    for (int nt = 0; nt < NT; ++nt) {
        int col = nt * 16 + mrow;
        float bv = bias[col];
        #pragma unroll
        for (int r = 0; r < 8; ++r) {
            float v = acc[nt][r] + bv;
            v = v > 0.f ? v : 0.f;
            long row = m0 + r + hi * 8;
            if (OUTH) outH[row * N + col] = (_Float16)v;
            else      outF[row * N + col] = v;
        }
    }
}

// ---------------------------------------------------------------------------
// Kernel C: sh_price from y1, y1_ ; inject into feature cols 79/80.
// ---------------------------------------------------------------------------
__global__ void shprice_kernel(const float* __restrict__ y1, const float* __restrict__ y1m,
                               const float* __restrict__ epw, const float* __restrict__ epb,
                               const float* __restrict__ ep1w, const float* __restrict__ ep1b,
                               const float* __restrict__ amp, const float* __restrict__ amp2,
                               const float* __restrict__ amp3,
                               _Float16* __restrict__ yA, _Float16* __restrict__ yB,
                               float* __restrict__ shp)
{
    long r = (long)blockIdx.x * blockDim.x + threadIdx.x;
    if (r >= BTOT) return;
    float e = 0.f, e1 = 0.f;
    #pragma unroll 8
    for (int k = 0; k < 64; ++k) {
        float s = y1[r * 64 + k] + y1m[r * 64 + k];
        e  += epw[k]  * s;
        e1 += ep1w[k] * s;
    }
    float sh0 = amp3[0] * tanhf(e + 2.f * epb[0]);
    float a1  = fabsf(e1 + 2.f * ep1b[0]);
    float sh1 = amp2[0] + amp[0] / (1.f + expf(-a1));
    shp[r * 2]     = sh0;
    shp[r * 2 + 1] = sh1;
    yA[r * 96 + 79] = (_Float16)sh0;  yA[r * 96 + 80] = (_Float16)sh1;
    yB[r * 96 + 79] = (_Float16)(-sh0); yB[r * 96 + 80] = (_Float16)sh1;
}

// ---------------------------------------------------------------------------
// Kernel E: pi heads + softmax combine -> open_0 ; cond -> inday.
// ---------------------------------------------------------------------------
__global__ void head_kernel(const float* __restrict__ y0, const float* __restrict__ y0m,
                            const float* __restrict__ piw, const float* __restrict__ pib,
                            const int* __restrict__ cat,
                            float* __restrict__ open0, float* __restrict__ inday,
                            float* __restrict__ outOpen)
{
    long r = (long)blockIdx.x * blockDim.x + threadIdx.x;
    if (r >= BTOT) return;
    float ls[3], ll[3];
    #pragma unroll
    for (int j = 0; j < 3; ++j) {
        float s = pib[j], t = pib[j];
        #pragma unroll 8
        for (int k = 0; k < 64; ++k) {
            float w = piw[j * 64 + k];
            s += w * y0m[r * 64 + k];
            t += w * y0 [r * 64 + k];
        }
        ls[j] = s; ll[j] = t;
    }
    float ms = fmaxf(ls[0], fmaxf(ls[1], ls[2]));
    float ml = fmaxf(ll[0], fmaxf(ll[1], ll[2]));
    float es0 = expf(ls[0] - ms), es1 = expf(ls[1] - ms), es2 = expf(ls[2] - ms);
    float el0 = expf(ll[0] - ml), el1 = expf(ll[1] - ml), el2 = expf(ll[2] - ml);
    float ssum = es0 + es1 + es2, lsum = el0 + el1 + el2;
    float s0 = es0 / ssum, s1 = es1 / ssum;
    float p0 = el0 / lsum, p1 = el1 / lsum;
    // open_long = flip(softmax); open_long[1:] = {p1, p0}
    float o0 = 0.5f * (s0 + p1);
    float o1 = 0.5f * (s1 + p0);
    float op = o1 - o0;

    int c0 = cat[r * 5 + 0], c1 = cat[r * 5 + 1], c2 = cat[r * 5 + 2],
        c3 = cat[r * 5 + 3], c4 = cat[r * 5 + 4];
    bool cond = (((c1 == 5) && (c0 != 4)) || ((c1 == 7) && (c0 == 4)))
                && (c2 == 11) && (c3 == 4);
    cond = cond || ((c0 + c1 + c2 + c3 + c4) == 0);
    float ind = cond ? 0.f : 1.f;

    open0[r] = op;
    inday[r] = ind;
    outOpen[r] = op;
}

// ---------------------------------------------------------------------------
// Kernel F: sequential 255-step scan; one wave per day-group (512 groups).
// Lane j carries slots j and j+32 of the 60-deep position/price FIFO.
// Shifts via __shfl_up; reductions via shfl_xor tree. No LDS, no barriers.
// ---------------------------------------------------------------------------
__global__ void scan_kernel(const float* __restrict__ price, const float* __restrict__ open0,
                            const float* __restrict__ shp, const float* __restrict__ inday,
                            float* __restrict__ outPos)
{
    int wave = threadIdx.x >> 5;
    int lane = threadIdx.x & 31;
    int n = blockIdx.x * (blockDim.x >> 5) + wave;
    if (n >= NGRP) return;

    float p0 = 0.f, p1 = 0.f;
    float q0x = 0.f, q0y = 0.f, q1x = 0.f, q1y = 0.f;
    float ef0 = fminf(1.f, 0.2f * (float)(HOLDN - 1 - lane));
    float ef1 = (lane < HOLDN - 32) ? fminf(1.f, 0.2f * (float)(HOLDN - 1 - (lane + 32))) : 0.f;
    long base = (long)n * DAYMAX;

    for (int t = 0; t < DAYMAX - 1; ++t) {
        long row = base + t;
        float rp  = price[row * 12];
        float op  = open0[row];
        float e0  = shp[row * 2];
        float e1  = shp[row * 2 + 1];
        float ind = inday[row];

        // exit_stop
        float es0 = (1.f - tanhf(fmaxf(0.f, fabsf(rp - q0x) - q0y))) * ef0;
        float es1 = (1.f - tanhf(fmaxf(0.f, fabsf(rp - q1x) - q1y))) * ef1;
        p0 *= es0; p1 *= es1;

        // head = relu(1 - sum|pos[:h-1]|) * op   (slot 59 excluded)
        float a = fabsf(p0) + ((lane < 27) ? fabsf(p1) : 0.f);
        #pragma unroll
        for (int off = 16; off; off >>= 1) a += __shfl_xor(a, off, 32);
        float head = fmaxf(0.f, 1.f - a) * op;

        // shift FIFO by one slot
        float p0prev = __shfl_up(p0, 1, 32);
        float p1prev = __shfl_up(p1, 1, 32);
        float p0l31  = __shfl(p0, 31, 32);
        float q0xp = __shfl_up(q0x, 1, 32), q0yp = __shfl_up(q0y, 1, 32);
        float q1xp = __shfl_up(q1x, 1, 32), q1yp = __shfl_up(q1y, 1, 32);
        float q0x31 = __shfl(q0x, 31, 32),  q0y31 = __shfl(q0y, 31, 32);

        float np0 = (lane == 0) ? head  : p0prev;
        float np1 = (lane == 0) ? p0l31 : p1prev;
        float nq0x = (lane == 0) ? (rp + e0) : q0xp;
        float nq0y = (lane == 0) ? e1        : q0yp;
        float nq1x = (lane == 0) ? q0x31 : q1xp;
        float nq1y = (lane == 0) ? q0y31 : q1yp;
        if (lane >= 28) { np1 = 0.f; nq1x = 0.f; nq1y = 0.f; } // slots >= 60 stay empty

        p0 = np0 * ind; p1 = np1 * ind;
        q0x = nq0x; q0y = nq0y; q1x = nq1x; q1y = nq1y;

        float s = p0 + p1;
        #pragma unroll
        for (int off = 16; off; off >>= 1) s += __shfl_xor(s, off, 32);
        if (lane == 0) outPos[row] = s;
    }
    if (lane == 0) outPos[base + DAYMAX - 1] = 0.f;
}

// ---------------------------------------------------------------------------
extern "C" void kernel_launch(void* const* d_in, const int* in_sizes, int n_in,
                              void* d_out, int out_size, void* d_ws, size_t ws_size,
                              hipStream_t stream)
{
    const float* x     = (const float*)d_in[0];
    const float* xp    = (const float*)d_in[1];
    const float* price = (const float*)d_in[2];
    const float* nom   = (const float*)d_in[3];
    const int*   cat   = (const int*)  d_in[4];
    // d_in[5] = day_max (256, hardcoded)
    const float* c00w  = (const float*)d_in[6];
    const float* c00b  = (const float*)d_in[7];
    const float* c10w  = (const float*)d_in[8];
    const float* c10b  = (const float*)d_in[9];
    const float* fc11w = (const float*)d_in[10];
    const float* fc11b = (const float*)d_in[11];
    const float* fc12w = (const float*)d_in[12];
    const float* fc12b = (const float*)d_in[13];
    const float* fc1ow = (const float*)d_in[14];
    const float* fc1ob = (const float*)d_in[15];
    const float* fc2ow = (const float*)d_in[16];
    const float* fc2ob = (const float*)d_in[17];
    const float* piw   = (const float*)d_in[18];
    const float* pib   = (const float*)d_in[19];
    const float* epw   = (const float*)d_in[20];
    const float* epb   = (const float*)d_in[21];
    const float* ep1w  = (const float*)d_in[22];
    const float* ep1b  = (const float*)d_in[23];
    const float* amp   = (const float*)d_in[24];
    const float* amp2  = (const float*)d_in[25];
    const float* amp3  = (const float*)d_in[26];
    const float* oh00  = (const float*)d_in[27];
    const float* oh01  = (const float*)d_in[28];
    const float* oh02  = (const float*)d_in[29];
    const float* oh03  = (const float*)d_in[30];
    const float* oh04  = (const float*)d_in[31];

    char* ws = (char*)d_ws;
    size_t off = 0;
    auto carve = [&](size_t bytes) -> void* {
        void* p = ws + off;
        off += (bytes + 255) & ~(size_t)255;
        return p;
    };
    _Float16* yA  = (_Float16*)carve((size_t)BTOT * 96 * 2);
    _Float16* yB  = (_Float16*)carve((size_t)BTOT * 96 * 2);
    _Float16* Hb  = (_Float16*)carve((size_t)BTOT * 128 * 2);
    float* y1  = (float*)carve((size_t)BTOT * 64 * 4);   // reused as y0
    float* y1m = (float*)carve((size_t)BTOT * 64 * 4);   // reused as y0_
    float* shp = (float*)carve((size_t)BTOT * 2 * 4);
    float* op0 = (float*)carve((size_t)BTOT * 4);
    float* ind = (float*)carve((size_t)BTOT * 4);

    float* outPos  = (float*)d_out;          // out_pos (B)
    float* outOpen = (float*)d_out + BTOT;   // open_0  (B)

    dim3 blk(256);
    dim3 gridElem(BTOT / 256);
    dim3 gridGemm(BTOT / 128);               // 8 waves x 16 rows per block

    features_kernel<<<gridElem, blk, 0, stream>>>(x, xp, price, nom, cat,
        c00w, c00b, c10w, c10b, oh00, oh01, oh02, oh03, oh04, yA, yB);

    // open_feature stack 1 (fc1o -> fc2o) on y and y_
    gemm_relu<96, 128, true ><<<gridGemm, blk, 0, stream>>>(yA, fc1ow, 79, fc1ob, Hb, nullptr);
    gemm_relu<128, 64, false><<<gridGemm, blk, 0, stream>>>(Hb, fc2ow, 128, fc2ob, nullptr, y1);
    gemm_relu<96, 128, true ><<<gridGemm, blk, 0, stream>>>(yB, fc1ow, 79, fc1ob, Hb, nullptr);
    gemm_relu<128, 64, false><<<gridGemm, blk, 0, stream>>>(Hb, fc2ow, 128, fc2ob, nullptr, y1m);

    shprice_kernel<<<gridElem, blk, 0, stream>>>(y1, y1m, epw, epb, ep1w, ep1b,
                                                 amp, amp2, amp3, yA, yB, shp);

    // open_feature stack 2 (fc11 -> fc12) on [y, sh_price] and [y_, sh_price*]
    gemm_relu<96, 128, true ><<<gridGemm, blk, 0, stream>>>(yA, fc11w, 81, fc11b, Hb, nullptr);
    gemm_relu<128, 64, false><<<gridGemm, blk, 0, stream>>>(Hb, fc12w, 128, fc12b, nullptr, y1);
    gemm_relu<96, 128, true ><<<gridGemm, blk, 0, stream>>>(yB, fc11w, 81, fc11b, Hb, nullptr);
    gemm_relu<128, 64, false><<<gridGemm, blk, 0, stream>>>(Hb, fc12w, 128, fc12b, nullptr, y1m);

    head_kernel<<<gridElem, blk, 0, stream>>>(y1, y1m, piw, pib, cat, op0, ind, outOpen);

    scan_kernel<<<NGRP / 8, blk, 0, stream>>>(price, op0, shp, ind, outPos);
}
